// HypergraphNeighborNet_80290118631450
// MI455X (gfx1250) — matrix operand order repaired and checked by reading, not compiled, not run on hardware
//
#include <hip/hip_runtime.h>
#include <hip/hip_bf16.h>
#include <cstdint>
#include <cstddef>

// ---------------------------------------------------------------------------
// HypergraphNeighborNet forward for MI455X (gfx1250), wave32.
// - All linears: tiled WMMA bf16 GEMM (f32 accumulate), operands delivered by
//   GLOBAL_LOAD_ASYNC_TO_LDS_B128 with double-buffered LDS (ASYNCcnt waits).
// - Weights converted fp32->bf16 once per launch; activation producers emit
//   bf16 directly, halving HBM traffic on the dominant edge-message GEMMs.
// - Attention kv-length == 1  =>  softmax == 1  =>  attend() == V projection.
// ---------------------------------------------------------------------------

#define N_NODES 10000
#define N_EDGES 30000
#define N_MOLS  256
#define ECHUNK  7680

typedef __bf16 bf16;
typedef __attribute__((ext_vector_type(16))) __bf16 bf16x16;
typedef __attribute__((ext_vector_type(8)))  float  floatx8;

__device__ __forceinline__ bf16 f2bf(float f) {
  uint32_t u = __builtin_bit_cast(uint32_t, f);
  u += 0x7fffu + ((u >> 16) & 1u);            // round-to-nearest-even
  uint16_t h = (uint16_t)(u >> 16);
  return __builtin_bit_cast(bf16, h);
}

__device__ __forceinline__ float gelu_exact(float x) {
  return 0.5f * x * (1.0f + erff(x * 0.7071067811865475f));
}

// ---- CDNA5 async global->LDS copy (16B per lane), ASYNCcnt-tracked --------
// VGLOBAL encoding, GVS mode: mem = SGPR64(base) + VGPR32(byte offset).
// First operand (VDST slot) is the VGPR holding the LDS byte address.
__device__ __forceinline__ void async_ld16(unsigned lds_off, unsigned goff,
                                           const void* base) {
  asm volatile("global_load_async_to_lds_b128 %0, %1, %2"
               :: "v"(lds_off), "v"(goff), "s"(base)
               : "memory");
}
__device__ __forceinline__ void wait_async4() {
  asm volatile("s_wait_asynccnt 0x4" ::: "memory");
}
__device__ __forceinline__ void wait_async0() {
  asm volatile("s_wait_asynccnt 0x0" ::: "memory");
}

#define TS 40  // LDS row stride in bf16: 80B rows -> every 16B chunk aligned

// C[M,N] = act(A[M,K] @ W[N,K]^T + bias); A,W bf16; C fp32 or bf16.
// 128 threads = 4 waves; block tile 64x64; wave = 2x2 WMMA 16x16x32 tiles.
// K tiles of 32 double-buffered through LDS via async copies.
template <int ACT, int CBF>
__global__ __launch_bounds__(128) void gemm_bf16(
    const bf16* __restrict__ A, int lda,
    const bf16* __restrict__ W,      // [N,K] row-major (reference 'w'), bf16
    const float* __restrict__ bias,  // [N] fp32
    void* __restrict__ Cv, int ldc,
    int M, int N, int K) {
  __shared__ bf16 At[2][64 * TS];
  __shared__ bf16 Bt[2][64 * TS];

  const int tid = threadIdx.x;
  const int m0 = blockIdx.y * 64;
  const int n0 = blockIdx.x * 64;

  // each thread owns 2 x 16B async slots: idx = tid*2+b -> row=idx>>2, k8=(idx&3)*8
  unsigned gaoff[2], gboff[2], loff[2];
#pragma unroll
  for (int b = 0; b < 2; ++b) {
    int idx = tid * 2 + b;
    int row = idx >> 2;
    int k8 = (idx & 3) * 8;
    int garow = m0 + row; if (garow >= M) garow = M - 1;
    int gwrow = n0 + row; if (gwrow >= N) gwrow = N - 1;
    gaoff[b] = (unsigned)(((size_t)garow * lda + k8) * sizeof(bf16));
    gboff[b] = (unsigned)(((size_t)gwrow * K + k8) * sizeof(bf16));
    loff[b] = (unsigned)((row * TS + k8) * sizeof(bf16));
  }
  const unsigned ldsA0 = (unsigned)(uintptr_t)&At[0][0];
  const unsigned ldsB0 = (unsigned)(uintptr_t)&Bt[0][0];
  const unsigned bufstride = (unsigned)(64 * TS * sizeof(bf16));

  const int lane = tid & 31;
  const int wv = tid >> 5;
  const int wm = (wv >> 1) * 32;
  const int wn = (wv & 1) * 32;
  const int r = lane & 15;
  const int half = lane >> 4;

  floatx8 acc[2][2];
#pragma unroll
  for (int i = 0; i < 2; ++i)
#pragma unroll
    for (int j = 0; j < 2; ++j) acc[i][j] = (floatx8)0.0f;

  auto issue = [&](int k0, int buf) {
    unsigned kb = (unsigned)(k0 * sizeof(bf16));
#pragma unroll
    for (int b = 0; b < 2; ++b) {
      async_ld16(ldsA0 + buf * bufstride + loff[b], gaoff[b] + kb, A);
      async_ld16(ldsB0 + buf * bufstride + loff[b], gboff[b] + kb, W);
    }
  };

  const int nt = K >> 5;  // all K dims are multiples of 32
  issue(0, 0);
  for (int t = 0; t < nt; ++t) {
    const int buf = t & 1;
    if (t + 1 < nt) { issue((t + 1) << 5, buf ^ 1); wait_async4(); }
    else            { wait_async0(); }
    __syncthreads();

    union Frag { bf16x16 v; uint32_t u[8]; };
    Frag af[2], bfr[2];
#pragma unroll
    for (int rb = 0; rb < 2; ++rb) {
      const uint32_t* ap = (const uint32_t*)(&At[buf][0] + (wm + rb * 16 + r) * TS);
#pragma unroll
      for (int q = 0; q < 8; ++q)           // ISA A 16x32 bf16 layout
        af[rb].u[q] = ap[(q & 3) + 4 * half + 8 * (q >> 2)];
    }
#pragma unroll
    for (int cb = 0; cb < 2; ++cb) {
      const uint32_t* bp = (const uint32_t*)(&Bt[buf][0] + (wn + cb * 16 + r) * TS);
#pragma unroll
      for (int q = 0; q < 8; ++q) bfr[cb].u[q] = bp[(q & 3) + 4 * half + 8 * (q >> 2)];  // same K order as A
    }
#pragma unroll
    for (int rb = 0; rb < 2; ++rb)
#pragma unroll
      for (int cb = 0; cb < 2; ++cb)
        acc[rb][cb] = __builtin_amdgcn_wmma_f32_16x16x32_bf16(
            false, af[rb].v, false, bfr[cb].v, (short)0, acc[rb][cb], false, false);
    __syncthreads();
  }

#pragma unroll
  for (int rb = 0; rb < 2; ++rb) {
#pragma unroll
    for (int cb = 0; cb < 2; ++cb) {
      const int gc = n0 + wn + cb * 16 + r;
      if (gc >= N) continue;
      const float bv = bias[gc];
#pragma unroll
      for (int i = 0; i < 8; ++i) {          // D: row = i + 8*half, col = lane&15
        const int gr = m0 + wm + rb * 16 + half * 8 + i;
        if (gr >= M) continue;
        float v = acc[rb][cb][i] + bv;
        if (ACT) v = gelu_exact(v);
        if (CBF) ((bf16*)Cv)[(size_t)gr * ldc + gc] = f2bf(v);
        else     ((float*)Cv)[(size_t)gr * ldc + gc] = v;
      }
    }
  }
}

// ---------------- element-wise / gather / scatter kernels -------------------

__global__ void f32_to_bf16_kernel(const float* __restrict__ src,
                                   bf16* __restrict__ dst, int n) {
  int i = blockIdx.x * blockDim.x + threadIdx.x;
  if (i < n) dst[i] = f2bf(src[i]);
}

__global__ void atom_embed_kernel(const int* __restrict__ atoms,
                                  const float* __restrict__ emb,
                                  bf16* __restrict__ XE, int n) {
  int i = blockIdx.x * blockDim.x + threadIdx.x;
  int node = i >> 7, j = i & 127;
  if (node >= n) return;
  XE[(size_t)node * 128 + j] = f2bf(emb[(size_t)atoms[node] * 128 + j]);
}

__global__ void edge_concat_kernel(const float* __restrict__ X,
                                   const float* __restrict__ edge_emb,
                                   const int* __restrict__ edges,
                                   const int* __restrict__ etyp,
                                   bf16* __restrict__ MIN, int e0, int ec, int E) {
  int e = blockIdx.x;
  if (e >= ec) return;
  int ge = e0 + e;
  int rr = edges[ge];
  int cc = edges[(size_t)E + ge];
  int tt = etyp[ge];
  bf16* dst = MIN + (size_t)e * 1088;
  for (int j = threadIdx.x; j < 512; j += blockDim.x) dst[j] = f2bf(X[(size_t)rr * 512 + j]);
  for (int j = threadIdx.x; j < 512; j += blockDim.x) dst[512 + j] = f2bf(X[(size_t)cc * 512 + j]);
  for (int j = threadIdx.x; j < 64; j += blockDim.x) dst[1024 + j] = f2bf(edge_emb[tt * 64 + j]);
}

__global__ void scatter_add_kernel(const float* __restrict__ MSG,
                                   const int* __restrict__ edges,
                                   float* __restrict__ AGGR, float* __restrict__ CNT,
                                   int e0, int ec, int E) {
  int e = blockIdx.x;
  if (e >= ec) return;
  int cc = edges[(size_t)E + e0 + e];
  const float* src = MSG + (size_t)e * 512;
  float* dst = AGGR + (size_t)cc * 512;
  for (int j = threadIdx.x; j < 512; j += blockDim.x) atomicAdd(dst + j, src[j]);
  if (threadIdx.x == 0) atomicAdd(CNT + cc, 1.0f);
}

__global__ void build_uin_kernel(const float* __restrict__ X,
                                 const float* __restrict__ AGGR,
                                 const float* __restrict__ CNT,
                                 bf16* __restrict__ UIN, int n) {
  int i = blockIdx.x * blockDim.x + threadIdx.x;
  int node = i >> 9, j = i & 511;
  if (node >= n) return;
  float c = CNT[node]; c = c > 1.0f ? c : 1.0f;
  UIN[(size_t)node * 1024 + j] = f2bf(X[(size_t)node * 512 + j]);
  UIN[(size_t)node * 1024 + 512 + j] = f2bf(AGGR[(size_t)node * 512 + j] / c);
}

// x = LN(x + upd) * g + b row-wise; optional bf16 mirror for GEMM consumers.
__global__ void add_ln_kernel(float* __restrict__ X, const float* __restrict__ UPD,
                              const float* __restrict__ g, const float* __restrict__ b,
                              int D, bf16* __restrict__ Xbf) {
  int row = blockIdx.x;
  __shared__ float red[256];
  float vals[4];
  int nv = 0;
  float s = 0.f;
  for (int j = threadIdx.x; j < D; j += 256) {
    float t = X[(size_t)row * D + j] + UPD[(size_t)row * D + j];
    vals[nv++] = t; s += t;
  }
  red[threadIdx.x] = s; __syncthreads();
  for (int st = 128; st > 0; st >>= 1) {
    if (threadIdx.x < st) red[threadIdx.x] += red[threadIdx.x + st];
    __syncthreads();
  }
  float mu = red[0] / (float)D; __syncthreads();
  float vs = 0.f;
  for (int k = 0; k < nv; ++k) { float d = vals[k] - mu; vs += d * d; }
  red[threadIdx.x] = vs; __syncthreads();
  for (int st = 128; st > 0; st >>= 1) {
    if (threadIdx.x < st) red[threadIdx.x] += red[threadIdx.x + st];
    __syncthreads();
  }
  float inv = rsqrtf(red[0] / (float)D + 1e-5f);
  int k = 0;
  for (int j = threadIdx.x; j < D; j += 256) {
    float o = (vals[k++] - mu) * inv * g[j] + b[j];
    X[(size_t)row * D + j] = o;
    if (Xbf) Xbf[(size_t)row * D + j] = f2bf(o);
  }
}

__global__ void pool_scatter_kernel(const float* __restrict__ X,
                                    const int* __restrict__ batch,
                                    float* __restrict__ S, float* __restrict__ C, int n) {
  int node = blockIdx.x;
  if (node >= n) return;
  int m = batch[node];
  for (int j = threadIdx.x; j < 512; j += blockDim.x)
    atomicAdd(S + (size_t)m * 512 + j, X[(size_t)node * 512 + j]);
  if (threadIdx.x == 0) atomicAdd(C + m, 1.0f);
}

__global__ void pool_mean_kernel(const float* __restrict__ S, const float* __restrict__ C,
                                 bf16* __restrict__ G) {
  int i = blockIdx.x * blockDim.x + threadIdx.x;
  if (i >= N_MOLS * 512) return;
  float c = C[i >> 9]; c = c > 1.0f ? c : 1.0f;
  G[i] = f2bf(S[i] / c);
}

// ---------------------------------------------------------------------------

extern "C" void kernel_launch(void* const* d_in, const int* in_sizes, int n_in,
                              void* d_out, int out_size, void* d_ws, size_t ws_size,
                              hipStream_t stream) {
  (void)in_sizes; (void)n_in; (void)out_size; (void)ws_size;
  auto F = [&](int i) { return (const float*)d_in[i]; };
  auto I = [&](int i) { return (const int*)d_in[i]; };

  // ---- params pytree leaf indices (sorted dict keys, 'b' < 'w') ----
  const int IDX_ATOM_EMB = 0, IDX_EDGE_EMB = 1, IDX_INP_B = 2, IDX_INP_W = 3;
  const int IDX_LAYER0 = 4;   // +10*l: ln.b, ln.g, m1.b, m1.w, m2.b, m2.w, u1.b, u1.w, u2.b, u2.w
  const int IDX_OUTP_B = 44, IDX_OUTP_W = 45;
  const int IDX_CO1_B = 46, IDX_CO1_W = 47, IDX_CO2_B = 48, IDX_CO2_W = 49;
  const int IDX_INPROJ_B = 50, IDX_INPROJ_W = 51;
  const int IDX_PROD1_B = 52, IDX_PROD1_W = 53, IDX_PROD2_B = 54, IDX_PROD2_W = 55;
  const int IDX_RXN1_B = 56, IDX_RXN1_W = 57, IDX_RXN2_B = 58, IDX_RXN2_W = 59;
  const int IDX_SC1_B = 60, IDX_SC1_W = 61, IDX_SC2_B = 62, IDX_SC2_W = 63;
  const int IDX_HYPER0 = 64;  // +22*l: f1.b,f1.w,f2.b,f2.w,ffln.b,ffln.g,k1.*,k2.*,ln.b,ln.g,out.b,out.w,q1.*,q2.*,v1.b,v1.w,v2.b,v2.w
  const int IDX_GRAPH0 = 196; // +4*g: atoms, edges, etypes, batch

  // ---- workspace carve-up (16B aligned) ----
  uintptr_t wp = (uintptr_t)d_ws;
  auto allocF = [&](size_t nf) { wp = (wp + 15) & ~(uintptr_t)15; float* p = (float*)wp; wp += nf * 4; return p; };
  auto allocH = [&](size_t nh) { wp = (wp + 15) & ~(uintptr_t)15; bf16* p = (bf16*)wp; wp += nh * 2; return p; };

  float* X    = allocF((size_t)N_NODES * 512);
  bf16*  XE   = allocH((size_t)N_NODES * 128);
  bf16*  MIN  = allocH((size_t)ECHUNK * 1088);
  bf16*  HIDB = allocH((size_t)ECHUNK * 1024);
  float* MSG  = allocF((size_t)ECHUNK * 512);
  float* AGGR = allocF((size_t)N_NODES * 512);
  float* CNT  = allocF((size_t)N_NODES);
  bf16*  UIN  = allocH((size_t)N_NODES * 1024);
  bf16*  UH   = allocH((size_t)N_NODES * 512);
  float* UPD  = allocF((size_t)N_NODES * 512);
  float* PS   = allocF((size_t)N_MOLS * 512);
  float* PC   = allocF((size_t)N_MOLS);
  bf16*  G1   = allocH((size_t)N_MOLS * 512);
  bf16*  G2   = allocH((size_t)N_MOLS * 512);
  float* HM   = allocF((size_t)N_MOLS * 768);
  bf16*  HMbf = allocH((size_t)N_MOLS * 768);
  bf16*  HPbf = allocH((size_t)N_MOLS * 768);
  bf16*  HCbf = allocH((size_t)N_MOLS * 768);
  bf16*  H12  = allocH((size_t)N_MOLS * 1536);
  float* UPDH = allocF((size_t)N_MOLS * 768);
  bf16*  FF   = allocH((size_t)N_MOLS * 3072);
  bf16*  T    = allocH((size_t)N_MOLS * 768);

  // bf16 weight mirrors
  bf16* Winp = allocH(512 * 128);
  bf16 *Wm1[4], *Wm2[4], *Wu1[4], *Wu2[4];
  for (int l = 0; l < 4; ++l) {
    Wm1[l] = allocH(1024 * 1088); Wm2[l] = allocH(512 * 1024);
    Wu1[l] = allocH(512 * 1024);  Wu2[l] = allocH(512 * 512);
  }
  bf16* Woutp = allocH(512 * 512);
  bf16* Winpj = allocH(768 * 512);
  bf16 *Wv1[6], *Wv2[6], *Wout[6], *Wf1[6], *Wf2[6];
  for (int l = 0; l < 6; ++l) {
    Wv1[l] = allocH(768 * 768); Wv2[l] = allocH(768 * 768);
    Wout[l] = allocH(768 * 1536);
    Wf1[l] = allocH(3072 * 768); Wf2[l] = allocH(768 * 3072);
  }
  bf16* Wprod1 = allocH(768 * 768); bf16* Wprod2 = allocH(512 * 768);
  bf16* Wco1   = allocH(768 * 768); bf16* Wco2   = allocH(512 * 768);
  bf16* Wrxn1  = allocH(384 * 768); bf16* Wrxn2  = allocH(10 * 384);
  bf16* Wsc1   = allocH(384 * 768); bf16* Wsc2   = allocH(1 * 384);

  auto cvt = [&](int pidx, bf16* dst, int n) {
    f32_to_bf16_kernel<<<(n + 255) / 256, 256, 0, stream>>>(F(pidx), dst, n);
  };
  cvt(IDX_INP_W, Winp, 512 * 128);
  for (int l = 0; l < 4; ++l) {
    int base = IDX_LAYER0 + 10 * l;
    cvt(base + 3, Wm1[l], 1024 * 1088);
    cvt(base + 5, Wm2[l], 512 * 1024);
    cvt(base + 7, Wu1[l], 512 * 1024);
    cvt(base + 9, Wu2[l], 512 * 512);
  }
  cvt(IDX_OUTP_W, Woutp, 512 * 512);
  cvt(IDX_INPROJ_W, Winpj, 768 * 512);
  for (int l = 0; l < 6; ++l) {
    int hb = IDX_HYPER0 + 22 * l;
    cvt(hb + 1, Wf1[l], 3072 * 768);
    cvt(hb + 3, Wf2[l], 768 * 3072);
    cvt(hb + 13, Wout[l], 768 * 1536);
    cvt(hb + 19, Wv1[l], 768 * 768);
    cvt(hb + 21, Wv2[l], 768 * 768);
  }
  cvt(IDX_PROD1_W, Wprod1, 768 * 768); cvt(IDX_PROD2_W, Wprod2, 512 * 768);
  cvt(IDX_CO1_W, Wco1, 768 * 768);     cvt(IDX_CO2_W, Wco2, 512 * 768);
  cvt(IDX_RXN1_W, Wrxn1, 384 * 768);   cvt(IDX_RXN2_W, Wrxn2, 10 * 384);
  cvt(IDX_SC1_W, Wsc1, 384 * 768);     cvt(IDX_SC2_W, Wsc2, 1 * 384);

  auto gemm = [&](const bf16* A, int lda, const bf16* W, const float* b,
                  void* C, int ldc, int M, int N, int K, bool act, bool cbf) {
    dim3 g((N + 63) / 64, (M + 63) / 64), blk(128);
    if (act) {
      if (cbf) gemm_bf16<1, 1><<<g, blk, 0, stream>>>(A, lda, W, b, C, ldc, M, N, K);
      else     gemm_bf16<1, 0><<<g, blk, 0, stream>>>(A, lda, W, b, C, ldc, M, N, K);
    } else {
      if (cbf) gemm_bf16<0, 1><<<g, blk, 0, stream>>>(A, lda, W, b, C, ldc, M, N, K);
      else     gemm_bf16<0, 0><<<g, blk, 0, stream>>>(A, lda, W, b, C, ldc, M, N, K);
    }
  };

  // ---- encoder (run on each of the 3 graphs) ----
  auto encode = [&](int g, void* Hout, bool houtBf) {
    const int* atoms = I(IDX_GRAPH0 + 4 * g + 0);
    const int* edges = I(IDX_GRAPH0 + 4 * g + 1);
    const int* etyp  = I(IDX_GRAPH0 + 4 * g + 2);
    const int* batch = I(IDX_GRAPH0 + 4 * g + 3);

    atom_embed_kernel<<<(N_NODES * 128 + 255) / 256, 256, 0, stream>>>(
        atoms, F(IDX_ATOM_EMB), XE, N_NODES);
    gemm(XE, 128, Winp, F(IDX_INP_B), X, 512, N_NODES, 512, 128, false, false);

    for (int l = 0; l < 4; ++l) {
      int base = IDX_LAYER0 + 10 * l;
      const float *lnb = F(base + 0), *lng = F(base + 1);
      const float *m1b = F(base + 2), *m2b = F(base + 4);
      const float *u1b = F(base + 6), *u2b = F(base + 8);

      (void)hipMemsetAsync(AGGR, 0, (size_t)N_NODES * 512 * sizeof(float), stream);
      (void)hipMemsetAsync(CNT, 0, (size_t)N_NODES * sizeof(float), stream);
      for (int e0 = 0; e0 < N_EDGES; e0 += ECHUNK) {
        int ec = N_EDGES - e0; if (ec > ECHUNK) ec = ECHUNK;
        edge_concat_kernel<<<ec, 128, 0, stream>>>(X, F(IDX_EDGE_EMB), edges, etyp,
                                                   MIN, e0, ec, N_EDGES);
        gemm(MIN, 1088, Wm1[l], m1b, HIDB, 1024, ec, 1024, 1088, true, true);   // gelu(m1)
        gemm(HIDB, 1024, Wm2[l], m2b, MSG, 512, ec, 512, 1024, false, false);   // m2
        scatter_add_kernel<<<ec, 128, 0, stream>>>(MSG, edges, AGGR, CNT, e0, ec, N_EDGES);
      }
      build_uin_kernel<<<(N_NODES * 512 + 255) / 256, 256, 0, stream>>>(X, AGGR, CNT, UIN, N_NODES);
      gemm(UIN, 1024, Wu1[l], u1b, UH, 512, N_NODES, 512, 1024, true, true);    // gelu(u1)
      gemm(UH, 512, Wu2[l], u2b, UPD, 512, N_NODES, 512, 512, false, false);    // u2
      add_ln_kernel<<<N_NODES, 256, 0, stream>>>(X, UPD, lng, lnb, 512, (bf16*)nullptr);
    }
    (void)hipMemsetAsync(PS, 0, (size_t)N_MOLS * 512 * sizeof(float), stream);
    (void)hipMemsetAsync(PC, 0, (size_t)N_MOLS * sizeof(float), stream);
    pool_scatter_kernel<<<N_NODES, 128, 0, stream>>>(X, batch, PS, PC, N_NODES);
    pool_mean_kernel<<<(N_MOLS * 512 + 255) / 256, 256, 0, stream>>>(PS, PC, G1);
    gemm(G1, 512, Woutp, F(IDX_OUTP_B), G2, 512, N_MOLS, 512, 512, false, true);
    gemm(G2, 512, Winpj, F(IDX_INPROJ_B), Hout, 768, N_MOLS, 768, 512, false, houtBf);
  };

  encode(0, HM, false);    // mol   (fp32: residual stream for LN)
  encode(1, HPbf, true);   // product (only ever a GEMM operand)
  encode(2, HCbf, true);   // co_react

  // seed HMbf (bf16 mirror of mol state) for first f1 GEMM
  f32_to_bf16_kernel<<<(N_MOLS * 768 + 255) / 256, 256, 0, stream>>>(HM, HMbf, N_MOLS * 768);

  // ---- hyper layers: kv length == 1 => attend() == V projection ----
  for (int l = 0; l < 6; ++l) {
    int hb = IDX_HYPER0 + 22 * l;
    const float *f1b = F(hb + 0), *f2b = F(hb + 2);
    const float *flnb = F(hb + 4), *flng = F(hb + 5);
    const float *lnb = F(hb + 10), *lng = F(hb + 11);
    const float *outb = F(hb + 12);
    const float *v1b = F(hb + 18), *v2b = F(hb + 20);

    gemm(HPbf, 768, Wv1[l], v1b, H12, 1536, N_MOLS, 768, 768, false, true);        // h1
    gemm(HCbf, 768, Wv2[l], v2b, H12 + 768, 1536, N_MOLS, 768, 768, false, true);  // h2
    gemm(H12, 1536, Wout[l], outb, UPDH, 768, N_MOLS, 768, 1536, false, false);    // out
    add_ln_kernel<<<N_MOLS, 256, 0, stream>>>(HM, UPDH, lng, lnb, 768, HMbf);
    gemm(HMbf, 768, Wf1[l], f1b, FF, 3072, N_MOLS, 3072, 768, true, true);         // gelu(f1)
    gemm(FF, 3072, Wf2[l], f2b, UPDH, 768, N_MOLS, 768, 3072, false, false);       // f2
    add_ln_kernel<<<N_MOLS, 256, 0, stream>>>(HM, UPDH, flng, flnb, 768, HMbf);
  }

  // ---- heads, written directly into d_out (fp32, concatenated) ----
  float* OUT = (float*)d_out;
  float* out_prod = OUT;                              // [256,512]
  float* out_co   = OUT + 256 * 512;                  // [256,512]
  float* out_rxn  = OUT + 2 * 256 * 512;              // [256,10]
  float* out_sc   = OUT + 2 * 256 * 512 + 256 * 10;   // [256,1]

  gemm(HMbf, 768, Wprod1, F(IDX_PROD1_B), T, 768, N_MOLS, 768, 768, true, true);
  gemm(T, 768, Wprod2, F(IDX_PROD2_B), out_prod, 512, N_MOLS, 512, 768, false, false);
  gemm(HMbf, 768, Wco1, F(IDX_CO1_B), T, 768, N_MOLS, 768, 768, true, true);
  gemm(T, 768, Wco2, F(IDX_CO2_B), out_co, 512, N_MOLS, 512, 768, false, false);
  gemm(HMbf, 768, Wrxn1, F(IDX_RXN1_B), T, 384, N_MOLS, 384, 768, true, true);
  gemm(T, 384, Wrxn2, F(IDX_RXN2_B), out_rxn, 10, N_MOLS, 10, 384, false, false);
  gemm(HMbf, 768, Wsc1, F(IDX_SC1_B), T, 384, N_MOLS, 384, 768, true, true);
  gemm(T, 384, Wsc2, F(IDX_SC2_B), out_sc, 1, N_MOLS, 1, 384, false, false);
}
